// PolicyNetwork_36232344109428
// MI455X (gfx1250) — matrix operand order repaired and checked by reading, not compiled, not run on hardware
//
#include <hip/hip_runtime.h>
#include <hip/hip_bf16.h>
#include <math.h>

// ---------------------------------------------------------------------------
// Policy network forward, fused for MI455X (gfx1250, wave32, WMMA).
//   B=256, OBS=128, H=4096, DEPTH=100, NOUT=64
// Memory-bound on the 840MB new_stack copy (~36us @ 23.3TB/s); GEMMs run in
// bf16 on the matrix pipe (v_wmma_f32_16x16x32_bf16), ~17.5 GFLOP total.
// ---------------------------------------------------------------------------

#define BATCH   256
#define OBS_N   128
#define HID     4096
#define DEPTH_N 100
#define NOUT_N  64
#define K1      (OBS_N + HID)          // 4224
#define K2      HID                    // 4096

// output layout (flat floats, reference return order)
#define LOGITS_OFF 0                          // 256*64 = 16384
#define VALUE_OFF  (BATCH * NOUT_N)           // 16384
#define STACK_OFF  (VALUE_OFF + BATCH)        // 16640
#define IDX_OFF    (STACK_OFF + BATCH * DEPTH_N * HID)  // 104874240

typedef __bf16 bf16;
typedef __attribute__((ext_vector_type(16))) __bf16 v16bf;
typedef __attribute__((ext_vector_type(8)))  __bf16 v8bf;
typedef __attribute__((ext_vector_type(8)))  float  v8f;
typedef __attribute__((ext_vector_type(4)))  float  v4f;   // native vector for NT builtins

union V16 { v16bf v; v8bf h[2]; };

// ---------------------------------------------------------------------------
// 1) Bulk copy stack -> new_stack (419MB each way). Nontemporal to keep L2
//    free for the W1/W2 weights (136MB, fits in 192MB L2).
// ---------------------------------------------------------------------------
__global__ __launch_bounds__(256)
void copy_stack_kernel(const v4f* __restrict__ src, v4f* __restrict__ dst, long n)
{
    long i = (long)blockIdx.x * blockDim.x + threadIdx.x;
    long stride = (long)gridDim.x * blockDim.x;
    for (; i < n; i += stride) {
        v4f v = __builtin_nontemporal_load(&src[i]);
        __builtin_nontemporal_store(v, &dst[i]);
    }
}

// ---------------------------------------------------------------------------
// 2) Build A = concat(x, stack[m, idx[m]]) as bf16 [256][4224]
// ---------------------------------------------------------------------------
__global__ __launch_bounds__(256)
void build_concat_kernel(const float* __restrict__ x,
                         const float* __restrict__ stack,
                         const int*   __restrict__ idx,
                         bf16* __restrict__ Abf)
{
    const int m = blockIdx.x;
    const float* top = stack + ((size_t)m * DEPTH_N + idx[m]) * HID;
    bf16* dst = Abf + (size_t)m * K1;
    for (int k = threadIdx.x; k < OBS_N; k += blockDim.x)
        dst[k] = (bf16)x[(size_t)m * OBS_N + k];
    for (int k = threadIdx.x; k < HID; k += blockDim.x)
        dst[OBS_N + k] = (bf16)top[k];
}

// ---------------------------------------------------------------------------
// 3) Tiled WMMA GEMM:  out[m][n] = tanh( sum_k A[m][k]*W[n][k] + bias[n] )
//    A: [256][K] bf16 (K-major), W: [4096][K] f32 (K-major) converted to bf16
//    while staged into LDS. Block tile 128x128, BK=32, 8 waves, each wave
//    computes 32(M) x 64(N) = 2x4 grid of 16x16 accumulators.
// ---------------------------------------------------------------------------
#define BM 128
#define BN 128
#define BK 32
#define LDP (BK + 8)   // LDS row pad: 80B rows keep 16B alignment, spread banks

template<int STORE_BF16>
__global__ __launch_bounds__(256)
void gemm_tanh_wmma_kernel(const bf16*  __restrict__ A, int K,
                           const float* __restrict__ W,
                           const float* __restrict__ bias,
                           bf16*  __restrict__ outBf,
                           float* __restrict__ outF)
{
    __shared__ bf16 As[BM][LDP];
    __shared__ bf16 Bs[BN][LDP];   // staged [n][k] (K-major, matches W rows)

    const int nBlock = blockIdx.x * BN;
    const int mBlock = blockIdx.y * BM;
    const int tid  = threadIdx.x;
    const int wave = tid >> 5;
    const int lane = tid & 31;
    const int wr = wave >> 1;      // 0..3 : M sub-tile (32 rows)
    const int wc = wave & 1;       // 0..1 : N sub-tile (64 cols)
    const int lm = lane & 15;
    const int lh = lane >> 4;

    v8f acc[2][4];
#pragma unroll
    for (int mi = 0; mi < 2; ++mi)
#pragma unroll
        for (int ni = 0; ni < 4; ++ni)
#pragma unroll
            for (int e = 0; e < 8; ++e) acc[mi][ni][e] = 0.0f;

    const int ldRow  = tid >> 1;          // 0..127
    const int ldHalf = (tid & 1) * 16;    // 0 or 16 (K offset)

    for (int kb = 0; kb < K; kb += BK) {
        // stage A tile (already bf16): 32B per thread, 2x 16B vectors
        {
            const bf16* src = A + (size_t)(mBlock + ldRow) * K + kb + ldHalf;
            *(v8bf*)&As[ldRow][ldHalf]     = *(const v8bf*)src;
            *(v8bf*)&As[ldRow][ldHalf + 8] = *(const v8bf*)(src + 8);
        }
        // stage W tile with f32->bf16 conversion: 64B per thread
        {
            const float* src = W + (size_t)(nBlock + ldRow) * K + kb + ldHalf;
#pragma unroll
            for (int i = 0; i < 16; i += 4) {
                v4f f = *(const v4f*)(src + i);
                Bs[ldRow][ldHalf + i + 0] = (bf16)f.x;
                Bs[ldRow][ldHalf + i + 1] = (bf16)f.y;
                Bs[ldRow][ldHalf + i + 2] = (bf16)f.z;
                Bs[ldRow][ldHalf + i + 3] = (bf16)f.w;
            }
        }
        __syncthreads();

        // per-lane fragments per ISA 7.12.2 (16-bit A 16x32, B 32x16 layouts)
        V16 afrag[2];
#pragma unroll
        for (int mi = 0; mi < 2; ++mi) {
            const bf16* p = &As[wr * 32 + mi * 16 + lm][0];
            afrag[mi].h[0] = *(const v8bf*)(p + lh * 8);        // K = lh*8 .. +7
            afrag[mi].h[1] = *(const v8bf*)(p + 16 + lh * 8);   // K = 16+lh*8 ..
        }
        V16 bfrag[4];
#pragma unroll
        for (int ni = 0; ni < 4; ++ni) {
            const bf16* p = &Bs[wc * 64 + ni * 16 + lm][0]; // col=lm, K=lh*8..+7, 16+lh*8..+7
            bfrag[ni].h[0] = *(const v8bf*)(p + lh * 8);
            bfrag[ni].h[1] = *(const v8bf*)(p + 16 + lh * 8);
        }

#pragma unroll
        for (int mi = 0; mi < 2; ++mi)
#pragma unroll
            for (int ni = 0; ni < 4; ++ni)
                acc[mi][ni] = __builtin_amdgcn_wmma_f32_16x16x32_bf16(
                    false, afrag[mi].v, false, bfrag[ni].v,
                    (short)0, acc[mi][ni], false, false);

        __syncthreads();
    }

    // epilogue: bias + tanh, write (C/D layout: col=lm, row=lh*8+v per tile)
#pragma unroll
    for (int mi = 0; mi < 2; ++mi)
#pragma unroll
        for (int ni = 0; ni < 4; ++ni) {
            const int cBase = nBlock + wc * 64 + ni * 16 + lm;
            const int rBase = mBlock + wr * 32 + mi * 16 + lh * 8;
            const float bv = bias[cBase];
#pragma unroll
            for (int v = 0; v < 8; ++v) {
                float val = tanhf(acc[mi][ni][v] + bv);
                const size_t o = (size_t)(rBase + v) * HID + cBase;
                if (STORE_BF16) outBf[o] = (bf16)val;
                else            outF[o]  = val;
            }
        }
}

// ---------------------------------------------------------------------------
// 4) Scatter p rows into new_stack[m][idx[m]+1]
// ---------------------------------------------------------------------------
__global__ __launch_bounds__(256)
void scatter_p_kernel(const float* __restrict__ P,
                      const int*   __restrict__ idx,
                      float* __restrict__ out_stack)
{
    const int m = blockIdx.x;
    float* dst = out_stack + ((size_t)m * DEPTH_N + idx[m] + 1) * HID;
    const float* src = P + (size_t)m * HID;
    for (int k = threadIdx.x; k < HID; k += blockDim.x)
        dst[k] = src[k];
}

// ---------------------------------------------------------------------------
// 5) Heads: 68 dot products of length 4096 per row; wave-per-output,
//    shuffle reduction; then logits/value/argmax-driven idx update.
// ---------------------------------------------------------------------------
__global__ __launch_bounds__(256)
void heads_kernel(const float* __restrict__ P,
                  const float* __restrict__ Wsh, const float* __restrict__ bsh,
                  const float* __restrict__ Wp,  const float* __restrict__ bp,
                  const float* __restrict__ Wv,  const float* __restrict__ bv,
                  const int*   __restrict__ stack_idx,
                  float* __restrict__ out)
{
    const int m = blockIdx.x;
    const int wave = threadIdx.x >> 5;
    const int lane = threadIdx.x & 31;
    __shared__ float head[68];

    const float* p = P + (size_t)m * HID;
    for (int j = wave; j < 68; j += 8) {
        const float* w = (j < 3) ? (Wsh + (size_t)j * HID)
                       : (j < 67) ? (Wp + (size_t)(j - 3) * HID)
                                  : Wv;
        float s = 0.0f;
        for (int k = lane; k < HID; k += 32)
            s = fmaf(p[k], w[k], s);
#pragma unroll
        for (int off = 16; off > 0; off >>= 1)
            s += __shfl_down(s, off, 32);
        if (lane == 0) head[j] = s;
    }
    __syncthreads();

    const int t = threadIdx.x;
    if (t < NOUT_N) {
        out[LOGITS_OFF + m * NOUT_N + t] = head[3 + t] + bp[t];
    } else if (t == 64) {
        out[VALUE_OFF + m] = head[67] + bv[0];
    } else if (t == 65) {
        // softmax is monotone -> argmax over raw logits (first-max tie break)
        const float l0 = head[0] + bsh[0];
        const float l1 = head[1] + bsh[1];
        const float l2 = head[2] + bsh[2];
        int op = 0; float best = l0;
        if (l1 > best) { best = l1; op = 1; }
        if (l2 > best) { op = 2; }
        int ni = stack_idx[m] + op - 1;
        if (ni < 0) ni = 0;
        out[IDX_OFF + m] = (float)ni;
    }
}

// ---------------------------------------------------------------------------
// Launch
// ---------------------------------------------------------------------------
extern "C" void kernel_launch(void* const* d_in, const int* in_sizes, int n_in,
                              void* d_out, int out_size, void* d_ws, size_t ws_size,
                              hipStream_t stream)
{
    const float* x     = (const float*)d_in[0];
    const float* stack = (const float*)d_in[1];
    const int*   sidx  = (const int*)  d_in[2];
    const float* W1    = (const float*)d_in[3];
    const float* b1    = (const float*)d_in[4];
    const float* W2    = (const float*)d_in[5];
    const float* b2    = (const float*)d_in[6];
    const float* Wsh   = (const float*)d_in[7];
    const float* bsh   = (const float*)d_in[8];
    const float* Wp    = (const float*)d_in[9];
    const float* bp    = (const float*)d_in[10];
    const float* Wv    = (const float*)d_in[11];
    const float* bv    = (const float*)d_in[12];
    float* out = (float*)d_out;

    // workspace: Abf [256][4224] bf16 | Hbf [256][4096] bf16 | P [256][4096] f32
    char* ws = (char*)d_ws;
    bf16*  Abf = (bf16*)ws;                               // 2,162,688 B
    bf16*  Hbf = (bf16*)(ws + 2162688);                   // 2,097,152 B
    float* P   = (float*)(ws + 2162688 + 2097152);        // 4,194,304 B (~8.1MB total)

    float* out_stack = out + STACK_OFF;

    // 1) bulk stack copy (memory-bound critical path)
    const long nvec4 = (long)BATCH * DEPTH_N * HID / 4;   // 26,214,400
    copy_stack_kernel<<<8192, 256, 0, stream>>>(
        (const v4f*)stack, (v4f*)out_stack, nvec4);

    // 2) gather + concat + convert
    build_concat_kernel<<<BATCH, 256, 0, stream>>>(x, stack, sidx, Abf);

    // 3) GEMM1: h = tanh([x,top] @ W1^T + b1), bf16 out
    gemm_tanh_wmma_kernel<1><<<dim3(HID / BN, BATCH / BM), 256, 0, stream>>>(
        Abf, K1, W1, b1, Hbf, nullptr);

    // 4) GEMM2: p = tanh(h @ W2^T + b2), f32 out
    gemm_tanh_wmma_kernel<0><<<dim3(HID / BN, BATCH / BM), 256, 0, stream>>>(
        Hbf, K2, W2, b2, nullptr, P);

    // 5) scatter p into new_stack[m][idx+1] (after the bulk copy, stream order)
    scatter_p_kernel<<<BATCH, 256, 0, stream>>>(P, sidx, out_stack);

    // 6) heads: logits, value, new_stack_idx
    heads_kernel<<<BATCH, 256, 0, stream>>>(
        P, Wsh, bsh, Wp, bp, Wv, bv, sidx, out);
}
